// PointNet2_seg_63419487093387
// MI455X (gfx1250) — matrix operand order repaired and checked by reading, not compiled, not run on hardware
//
#include <hip/hip_runtime.h>
#include <hip/hip_bf16.h>

// ---------------------------------------------------------------------------
// PointNet++-style segmentation block for MI455X (gfx1250, wave32, WMMA).
// Per-point neighborhood MLPs as f16 V_WMMA_F32_16X16X32_F16 GEMMs, valid-
// neighbor tile skipping (ball query compaction), bias seeded into the WMMA
// C operand, leaky-ReLU as max(x, 0.2x), contiguous vectorizable gathers.
// ---------------------------------------------------------------------------

constexpr int BATCH = 2;
constexpr int NPTS  = 1024;
constexpr int KNN   = 512;
constexpr int FIN0  = 16;   // input feature channels
constexpr int OCH   = 64;   // hidden/output channels
constexpr float RAD2 = 1.0f;

constexpr int KPAD0 = 32;   // (16+3) -> 32
constexpr int KPAD1 = 96;   // (64+3) -> 96

typedef __attribute__((ext_vector_type(8)))  _Float16 v8h;
typedef __attribute__((ext_vector_type(16))) _Float16 v16h;
typedef __attribute__((ext_vector_type(8)))  float    v8f;

// ---------------------------------------------------------------------------
// Fragment loaders (wave32 CDNA5 WMMA layouts, 16-bit operands, K=32 step).
// A (16xK, M-rows): lanes 0-15 row M=lane, K = kb+{0..7, 16..23};
//                   lanes 16-31 row M=lane-16, K = kb+{8..15, 24..31}.
// B (Kx16, N-cols, stored col-major wt[c][k]): lanes 0-15 col N=lane holding
//                   K = kb+{0..7, 16..23}; lanes 16-31 col N=lane-16, K = kb+{8..15, 24..31}.
// ---------------------------------------------------------------------------
__device__ __forceinline__ v16h cat8(v8h lo, v8h hi) {
  return __builtin_shufflevector(lo, hi, 0, 1, 2, 3, 4, 5, 6, 7,
                                          8, 9, 10, 11, 12, 13, 14, 15);
}

__device__ __forceinline__ v16h load_frag_a(const _Float16* tile, int ld, int kb) {
  const int lane = threadIdx.x & 31;
  const int row  = lane & 15;
  const int koff = kb + ((lane & 16) ? 8 : 0);
  const _Float16* p = tile + row * ld + koff;
  v8h lo = *(const v8h*)(p);        // 16B aligned (ld multiple of 8, kb of 32)
  v8h hi = *(const v8h*)(p + 16);
  return cat8(lo, hi);
}

__device__ __forceinline__ v16h load_frag_b(const _Float16* tile, int ld, int kb) {
  const int lane = threadIdx.x & 31;
  const int col  = lane & 15;
  const int koff = kb + ((lane & 16) ? 8 : 0);
  const _Float16* p = tile + col * ld + koff;
  v8h lo = *(const v8h*)(p);
  v8h hi = *(const v8h*)(p + 16);
  return cat8(lo, hi);
}

__device__ __forceinline__ float leaky02(float x) {
  return fmaxf(x, 0.2f * x);        // slope in (0,1): exact leaky-ReLU
}

// ---------------------------------------------------------------------------
// Kernel 1: ball query (first-KNN in index order within radius) + valid count.
// One wave32 per query point; ballot/popcount compaction preserves index order.
// ---------------------------------------------------------------------------
__global__ __launch_bounds__(256) void ball_query_kernel(
    const float* __restrict__ pos, unsigned short* __restrict__ idx,
    int* __restrict__ cnt) {
  const int lane = threadIdx.x & 31;
  const int wid  = threadIdx.x >> 5;
  const int pt   = blockIdx.x * 8 + wid;           // [0, BATCH*NPTS)
  const int b = pt / NPTS, n = pt - b * NPTS;
  const float* pb = pos + (size_t)b * NPTS * 3;
  const float qx = pb[n * 3 + 0];
  const float qy = pb[n * 3 + 1];
  const float qz = pb[n * 3 + 2];

  int count = 0;
  for (int base = 0; base < NPTS && count < KNN; base += 32) {
    const int j = base + lane;                     // NPTS % 32 == 0
    const float dx = pb[j * 3 + 0] - qx;
    const float dy = pb[j * 3 + 1] - qy;
    const float dz = pb[j * 3 + 2] - qz;
    const bool in = (dx * dx + dy * dy + dz * dz) < RAD2;
    const unsigned m = (unsigned)__ballot(in);
    const int p = count + __popc(m & ((1u << lane) - 1u));
    if (in && p < KNN) idx[(size_t)pt * KNN + p] = (unsigned short)j;
    count += __popc(m);
  }
  if (count > KNN) count = KNN;
  if (lane == 0) cnt[pt] = count;
}

// ---------------------------------------------------------------------------
// Kernel 2: weight prepack — transpose to col-major f16, zero-pad K to KPAD.
// wt[c * KPAD + k] = (k < KIN) ? w[k * 64 + c] : 0
// ---------------------------------------------------------------------------
__device__ __forceinline__ void pack_one(const float* __restrict__ w,
                                         _Float16* __restrict__ wt,
                                         int kin, int kpad, int tid, int nth) {
  for (int e = tid; e < OCH * kpad; e += nth) {
    const int c = e / kpad, k = e - c * kpad;
    wt[e] = (k < kin) ? (_Float16)w[k * OCH + c] : (_Float16)0.0f;
  }
}

__global__ __launch_bounds__(256) void prepack_kernel(
    const float* __restrict__ w01, const float* __restrict__ w02,
    const float* __restrict__ w11, const float* __restrict__ w12,
    _Float16* __restrict__ wt01, _Float16* __restrict__ wt02,
    _Float16* __restrict__ wt11, _Float16* __restrict__ wt12) {
  const int tid = blockIdx.x * blockDim.x + threadIdx.x;
  const int nth = gridDim.x * blockDim.x;
  pack_one(w01, wt01, FIN0 + 3, KPAD0, tid, nth);
  pack_one(w02, wt02, OCH,      OCH,   tid, nth);
  pack_one(w11, wt11, OCH + 3,  KPAD1, tid, nth);
  pack_one(w12, wt12, OCH,      OCH,   tid, nth);
}

// ---------------------------------------------------------------------------
// Kernel 3: fused SA module + post-MLP + linear skip. One block per point.
// 8 waves; each wave processes 16-neighbor tiles with WMMA f16 GEMMs.
// ---------------------------------------------------------------------------
template <int CIN, int KPAD>
__global__ __launch_bounds__(256) void sa_layer_kernel(
    const float* __restrict__ pos,            // [B,N,3]
    const float* __restrict__ fin,            // [B,N,CIN]
    const unsigned short* __restrict__ idx,   // [B*N, KNN]
    const int* __restrict__ cntp,             // [B*N]
    const _Float16* __restrict__ wt1,         // [64][KPAD] col-major f16
    const float* __restrict__ b1,
    const _Float16* __restrict__ wt2,         // [64][64]  col-major f16
    const float* __restrict__ b2,
    const float* __restrict__ lw1, const float* __restrict__ lb1,
    const float* __restrict__ lw2, const float* __restrict__ lb2,
    const float* __restrict__ ltw,            // [CIN][64]
    const float* __restrict__ ltb,
    float* __restrict__ out)                  // [B,N,64]
{
  constexpr int NW = 8;
  constexpr int PADROW = (KPAD > 64) ? KPAD : 64;   // A (ld=KPAD) / h1 (ld=64) overlay
  constexpr int KH = KPAD / 2;                      // per-lane contiguous half-row

  __shared__ __align__(16) _Float16 s_wt1[OCH * KPAD];
  __shared__ __align__(16) _Float16 s_wt2[OCH * OCH];
  __shared__ __align__(16) _Float16 s_pad[NW][16 * PADROW];
  __shared__ float s_b1[OCH], s_b2[OCH];
  __shared__ float s_cmax[NW][OCH];
  __shared__ float s_vec[OCH];
  __shared__ float s_vec2[OCH];

  const int tid  = threadIdx.x;
  const int lane = tid & 31;
  const int wid  = tid >> 5;
  const int pt   = blockIdx.x;
  const int b = pt / NPTS, n = pt - b * NPTS;

  for (int e = tid; e < OCH * KPAD; e += 256) s_wt1[e] = wt1[e];
  for (int e = tid; e < OCH * OCH;  e += 256) s_wt2[e] = wt2[e];
  if (tid < OCH) { s_b1[tid] = b1[tid]; s_b2[tid] = b2[tid]; }

  const int   c_n    = cntp[pt];
  const int   ntiles = (c_n + 15) >> 4;
  const float initv  = (c_n < KNN) ? 0.0f : -3.4e38f;  // masked slots contribute 0

  const float qx = pos[(b * NPTS + n) * 3 + 0];
  const float qy = pos[(b * NPTS + n) * 3 + 1];
  const float qz = pos[(b * NPTS + n) * 3 + 2];

  const int col   = lane & 15;
  const int rbase = (lane & 16) ? 8 : 0;
  float wmax[4] = {-3.4e38f, -3.4e38f, -3.4e38f, -3.4e38f};

  __syncthreads();  // weights + biases resident

  // Broadcast biases into WMMA C-operand layout (lane owns one column; all 8
  // accumulator slots of that lane are rows of the same column -> same bias).
  v8f cb1[4], cb2[4];
#pragma unroll
  for (int nt = 0; nt < 4; ++nt) {
    const float bb1 = s_b1[nt * 16 + col];
    const float bb2 = s_b2[nt * 16 + col];
#pragma unroll
    for (int i = 0; i < 8; ++i) { cb1[nt][i] = bb1; cb2[nt][i] = bb2; }
  }

  for (int t = wid; t < ntiles; t += NW) {
    const int rows_valid = min(16, c_n - t * 16);
    _Float16* A = &s_pad[wid][0];

    // ---- gather [relxyz | feats] tile, f32 -> f16, zero-padded ----
    // Two lanes per row, each owning a CONTIGUOUS half-row -> vectorizable
    // global feature loads and coalesced LDS stores.
    {
      const int row  = lane & 15;
      const int part = lane >> 4;
      const int gr   = t * 16 + row;
      const bool v   = row < rows_valid;
      const int  j   = v ? (int)idx[(size_t)pt * KNN + gr] : 0;
      float rx = 0.f, ry = 0.f, rz = 0.f;
      if (v) {
        rx = pos[(b * NPTS + j) * 3 + 0] - qx;
        ry = pos[(b * NPTS + j) * 3 + 1] - qy;
        rz = pos[(b * NPTS + j) * 3 + 2] - qz;
      }
      const float* fj = fin + ((size_t)b * NPTS + j) * CIN;
      _Float16* Ar = A + row * KPAD + part * KH;
#pragma unroll
      for (int kk = 0; kk < KH; ++kk) {
        const int k = part * KH + kk;
        float val = 0.f;
        if (v) {
          if (k == 0) val = rx;
          else if (k == 1) val = ry;
          else if (k == 2) val = rz;
          else if (k - 3 < CIN) val = fj[k - 3];
        }
        Ar[kk] = (_Float16)val;
      }
    }

    // ---- GEMM1: [16 x KPAD] @ [KPAD x 64], bias pre-seeded in C ----
    v8f acc[4] = {cb1[0], cb1[1], cb1[2], cb1[3]};
    for (int ks = 0; ks < KPAD; ks += 32) {
      const v16h af = load_frag_a(A, KPAD, ks);
#pragma unroll
      for (int nt = 0; nt < 4; ++nt) {
        const v16h bf = load_frag_b(&s_wt1[(nt * 16) * KPAD], KPAD, ks);
        acc[nt] = __builtin_amdgcn_wmma_f32_16x16x32_f16(
            false, af, false, bf, (short)0, acc[nt], false, false);
      }
    }

    // ---- leaky, emit h1 (f16) into the same LDS scratch (ld=64) ----
    _Float16* H = &s_pad[wid][0];
#pragma unroll
    for (int nt = 0; nt < 4; ++nt) {
#pragma unroll
      for (int r = 0; r < 8; ++r) {
        H[(r + rbase) * OCH + (nt * 16 + col)] = (_Float16)leaky02(acc[nt][r]);
      }
    }

    // ---- GEMM2: [16 x 64] @ [64 x 64], bias pre-seeded in C ----
    v8f acc2[4] = {cb2[0], cb2[1], cb2[2], cb2[3]};
    for (int ks = 0; ks < OCH; ks += 32) {
      const v16h af = load_frag_a(H, OCH, ks);
#pragma unroll
      for (int nt = 0; nt < 4; ++nt) {
        const v16h bf = load_frag_b(&s_wt2[(nt * 16) * OCH], OCH, ks);
        acc2[nt] = __builtin_amdgcn_wmma_f32_16x16x32_f16(
            false, af, false, bf, (short)0, acc2[nt], false, false);
      }
    }

    // ---- leaky + row mask, fold column max into registers ----
#pragma unroll
    for (int nt = 0; nt < 4; ++nt) {
      float m = -3.4e38f;
#pragma unroll
      for (int r = 0; r < 8; ++r) {
        float x = leaky02(acc2[nt][r]);
        x = ((r + rbase) < rows_valid) ? x : 0.0f;   // mask padded rows
        m = fmaxf(m, x);
      }
      m = fmaxf(m, __shfl_xor(m, 16, 32));           // fold M=0..7 with M=8..15
      wmax[nt] = fmaxf(wmax[nt], m);
    }
  }

  if (lane < 16) {
#pragma unroll
    for (int nt = 0; nt < 4; ++nt) s_cmax[wid][nt * 16 + lane] = wmax[nt];
  }
  __syncthreads();

  // ---- cross-wave max, then post-MLP + linear skip on 64 lanes ----
  if (tid < OCH) {
    float xi = initv;
#pragma unroll
    for (int w = 0; w < NW; ++w) xi = fmaxf(xi, s_cmax[w][tid]);
    s_vec[tid] = xi;
  }
  __syncthreads();
  if (tid < OCH) {
    float a = lb1[tid];
    for (int d = 0; d < OCH; ++d) a += s_vec[d] * lw1[d * OCH + tid];
    s_vec2[tid] = (a > 0.f) ? a : 0.f;
  }
  __syncthreads();
  if (tid < OCH) {
    float a = lb2[tid];
    for (int d = 0; d < OCH; ++d) a += s_vec2[d] * lw2[d * OCH + tid];
    float sk = ltb[tid];
    const float* f0 = fin + (size_t)pt * CIN;
    for (int d = 0; d < CIN; ++d) sk += f0[d] * ltw[d * OCH + tid];
    out[(size_t)pt * OCH + tid] = sk + a;
  }
}

// ---------------------------------------------------------------------------
// Launch: ball query -> prepack -> layer0 (CIN=16,KPAD=32) -> layer1 (64,96)
// ---------------------------------------------------------------------------
extern "C" void kernel_launch(void* const* d_in, const int* in_sizes, int n_in,
                              void* d_out, int out_size, void* d_ws, size_t ws_size,
                              hipStream_t stream) {
  (void)in_sizes; (void)n_in; (void)out_size; (void)ws_size;

  const float* pos     = (const float*)d_in[0];
  const float* feats   = (const float*)d_in[1];
  const float* sa0_w1  = (const float*)d_in[2];
  const float* sa0_b1  = (const float*)d_in[3];
  const float* sa0_w2  = (const float*)d_in[4];
  const float* sa0_b2  = (const float*)d_in[5];
  const float* sa1_w1  = (const float*)d_in[6];
  const float* sa1_b1  = (const float*)d_in[7];
  const float* sa1_w2  = (const float*)d_in[8];
  const float* sa1_b2  = (const float*)d_in[9];
  const float* lin0_w1 = (const float*)d_in[10];
  const float* lin0_b1 = (const float*)d_in[11];
  const float* lin0_w2 = (const float*)d_in[12];
  const float* lin0_b2 = (const float*)d_in[13];
  const float* lin1_w1 = (const float*)d_in[14];
  const float* lin1_b1 = (const float*)d_in[15];
  const float* lin1_w2 = (const float*)d_in[16];
  const float* lin1_b2 = (const float*)d_in[17];
  const float* lt0_w   = (const float*)d_in[18];
  const float* lt0_b   = (const float*)d_in[19];
  const float* lt1_w   = (const float*)d_in[20];
  const float* lt1_b   = (const float*)d_in[21];

  unsigned char* ws = (unsigned char*)d_ws;
  size_t off = 0;
  unsigned short* idx = (unsigned short*)(ws + off); off += (size_t)BATCH * NPTS * KNN * 2;
  int*      cntp  = (int*)(ws + off);       off += (size_t)BATCH * NPTS * 4;
  _Float16* wt01  = (_Float16*)(ws + off);  off += (size_t)OCH * KPAD0 * 2;
  _Float16* wt02  = (_Float16*)(ws + off);  off += (size_t)OCH * OCH * 2;
  _Float16* wt11  = (_Float16*)(ws + off);  off += (size_t)OCH * KPAD1 * 2;
  _Float16* wt12  = (_Float16*)(ws + off);  off += (size_t)OCH * OCH * 2;
  float*    x0    = (float*)(ws + off);     off += (size_t)BATCH * NPTS * OCH * 4;

  ball_query_kernel<<<(BATCH * NPTS) / 8, 256, 0, stream>>>(pos, idx, cntp);
  prepack_kernel<<<32, 256, 0, stream>>>(sa0_w1, sa0_w2, sa1_w1, sa1_w2,
                                         wt01, wt02, wt11, wt12);
  sa_layer_kernel<FIN0, KPAD0><<<BATCH * NPTS, 256, 0, stream>>>(
      pos, feats, idx, cntp, wt01, sa0_b1, wt02, sa0_b2,
      lin0_w1, lin0_b1, lin0_w2, lin0_b2, lt0_w, lt0_b, x0);
  sa_layer_kernel<OCH, KPAD1><<<BATCH * NPTS, 256, 0, stream>>>(
      pos, x0, idx, cntp, wt11, sa1_b1, wt12, sa1_b2,
      lin1_w1, lin1_b1, lin1_w2, lin1_b2, lt1_w, lt1_b, (float*)d_out);
}
